// UMABlock_30176440222433
// MI455X (gfx1250) — matrix operand order repaired and checked by reading, not compiled, not run on hardware
//
#include <hip/hip_runtime.h>

// ---------------------------------------------------------------------------
// UMA-style equivariant GNN forward for MI455X (gfx1250, wave32, WMMA).
// All dense GEMMs (~260 GFLOP total) run on v_wmma_f32_16x16x32_f16 with
// f32 accumulate. Weight (B) tiles are staged into LDS with CDNA5 async
// global->LDS copies (ASYNCcnt) using double buffering, so the DMA overlaps
// the WMMA stream. Weights are pre-transposed once so both sides of the
// copy are contiguous. Irregular work (rotations, RBF, LN, gather/scatter)
// is wave32 VALU; segment_sum uses global f32 atomics.
// ---------------------------------------------------------------------------

#define N_NODES_   10000
#define N_EDGES_   60000
#define C_DIM      128
#define H_DIM      128
#define EC_DIM     128
#define NRBF_      512
#define EIN_       (NRBF_ + 2*EC_DIM)   // 768
#define CUTOFF_F   5.0f
#define NLAYERS_   2
#define RESCALE_F  5.0f
#define NGRID_     72

typedef __attribute__((ext_vector_type(16))) _Float16 v16h;
typedef __attribute__((ext_vector_type(8)))  _Float16 v8h;
typedef __attribute__((ext_vector_type(8)))  float    v8f;

// Normalized A2 basis (5 symmetric traceless 3x3 matrices), row-major.
__constant__ float A2c[5][9] = {
  {0.f, 0.70710678f, 0.f,   0.70710678f, 0.f, 0.f,   0.f, 0.f, 0.f},
  {0.f, 0.f, 0.f,   0.f, 0.f, 0.70710678f,   0.f, 0.70710678f, 0.f},
  {-0.40824829f, 0.f, 0.f,   0.f, -0.40824829f, 0.f,   0.f, 0.f, 0.81649658f},
  {0.f, 0.f, 0.70710678f,   0.f, 0.f, 0.f,   0.70710678f, 0.f, 0.f},
  {0.70710678f, 0.f, 0.f,   0.f, -0.70710678f, 0.f,   0.f, 0.f, 0.f}};

// ---------------------------------------------------------------------------
// fp32 (K,N) -> fp16 transposed (N,K) weight conversion (done once; weights
// are tiny, so this makes every later B-tile stage a contiguous copy).
// ---------------------------------------------------------------------------
__global__ void k_transpose_f16(const float* __restrict__ W, _Float16* __restrict__ Wt,
                                int K, int N)
{
  int i = blockIdx.x * 256 + threadIdx.x;
  if (i < K * N) {
    int k = i / N, n = i % N;
    Wt[(size_t)n * K + k] = (_Float16)W[i];
  }
}

// ---------------------------------------------------------------------------
// Build TO_GRID (72x9) and FROM_GRID = pinv (9x72) on device.
// ---------------------------------------------------------------------------
__global__ void k_make_grid(float* __restrict__ TG, float* __restrict__ FG)
{
  __shared__ float G[NGRID_][9];
  __shared__ float Maug[9][18];
  const float ct[6] = {-0.9324695142031521f, -0.6612093864662645f, -0.2386191860831969f,
                        0.2386191860831969f,  0.6612093864662645f,  0.9324695142031521f};
  int t = threadIdx.x;
  if (t < NGRID_) {
    int i = t / 12, j = t % 12;
    float z  = ct[i];
    float st = sqrtf(fmaxf(1.f - z*z, 0.f));
    float ph = (2.f * 3.14159265358979323846f / 12.f) * (float)j;
    float x = st * cosf(ph), y = st * sinf(ph);
    float sh9[9] = {0.28209479177f,
                    0.4886025119f * y, 0.4886025119f * z, 0.4886025119f * x,
                    1.0925484306f * x * y, 1.0925484306f * y * z,
                    0.3153915653f * (3.f*z*z - 1.f),
                    1.0925484306f * x * z, 0.5462742153f * (x*x - y*y)};
    for (int l = 0; l < 9; ++l) { G[t][l] = sh9[l]; TG[t*9 + l] = sh9[l]; }
  }
  __syncthreads();
  if (t < 81) {
    int i = t / 9, j = t % 9;
    float s = 0.f;
    for (int p = 0; p < NGRID_; ++p) s += G[p][i] * G[p][j];
    Maug[i][j] = s;
    Maug[i][9 + j] = (i == j) ? 1.f : 0.f;
  }
  __syncthreads();
  if (t == 0) {
    for (int col = 0; col < 9; ++col) {
      int piv = col; float best = fabsf(Maug[col][col]);
      for (int r = col + 1; r < 9; ++r) {
        float a = fabsf(Maug[r][col]);
        if (a > best) { best = a; piv = r; }
      }
      if (piv != col)
        for (int j = 0; j < 18; ++j) { float tmp = Maug[col][j]; Maug[col][j] = Maug[piv][j]; Maug[piv][j] = tmp; }
      float ipv = 1.f / Maug[col][col];
      for (int j = 0; j < 18; ++j) Maug[col][j] *= ipv;
      for (int r = 0; r < 9; ++r) if (r != col) {
        float f = Maug[r][col];
        for (int j = 0; j < 18; ++j) Maug[r][j] -= f * Maug[col][j];
      }
    }
  }
  __syncthreads();
  for (int idx = t; idx < 9 * NGRID_; idx += blockDim.x) {
    int l = idx / NGRID_, p = idx % NGRID_;
    float s = 0.f;
    for (int i = 0; i < 9; ++i) s += Maug[l][9 + i] * G[p][i];
    FG[idx] = s;
  }
}

// ---------------------------------------------------------------------------
// Per-edge geometry: distance, envelope, D1 (3x3), D2 (5x5)
// ---------------------------------------------------------------------------
__global__ void k_edge_geom(const float* __restrict__ ev, const float* __restrict__ gamma,
                            float* __restrict__ D1g, float* __restrict__ D2g,
                            float* __restrict__ env, float* __restrict__ dist, int E)
{
  int e = blockIdx.x * blockDim.x + threadIdx.x;
  if (e >= E) return;
  float vx = ev[e*3+0], vy = ev[e*3+1], vz = ev[e*3+2];
  float d = sqrtf(vx*vx + vy*vy + vz*vz);
  dist[e] = d;
  float x1 = d / CUTOFF_F;
  float x2 = x1*x1, x4 = x2*x2, x5 = x4*x1, x6 = x5*x1, x7 = x6*x1;
  float en = 1.f - 21.f*x5 + 35.f*x6 - 15.f*x7;
  env[e] = (x1 < 1.f) ? en : 0.f;
  float inv = 1.f / (d + 1e-12f);
  float nx = vx*inv, ny = vy*inv, nz = vz*inv;
  float ux, uy, uz;
  if (fabsf(nx) > 0.9f) { ux = 0.f; uy = 1.f; uz = 0.f; }
  else                  { ux = 1.f; uy = 0.f; uz = 0.f; }
  float dot = ux*nx + uy*ny + uz*nz;
  float tx = ux - dot*nx, ty = uy - dot*ny, tz = uz - dot*nz;
  float tn = 1.f / (sqrtf(tx*tx + ty*ty + tz*tz) + 1e-12f);
  float b1x = tx*tn, b1y = ty*tn, b1z = tz*tn;
  float b3x = b1y*nz - b1z*ny;
  float b3y = b1z*nx - b1x*nz;
  float b3z = b1x*ny - b1y*nx;
  float R0[3][3] = {{b1x,b1y,b1z},{nx,ny,nz},{b3x,b3y,b3z}};
  float g = gamma[e]; float cg = cosf(g), sg = sinf(g);
  float Rg[3][3] = {{cg,0.f,sg},{0.f,1.f,0.f},{-sg,0.f,cg}};
  float R[3][3];
  for (int i = 0; i < 3; ++i)
    for (int j = 0; j < 3; ++j) {
      float s = 0.f;
      for (int k = 0; k < 3; ++k) s += Rg[i][k] * R0[k][j];
      R[i][j] = s;
    }
  const int perm[3] = {1, 2, 0};
  for (int i = 0; i < 3; ++i)
    for (int j = 0; j < 3; ++j)
      D1g[(size_t)e*9 + i*3 + j] = R[perm[i]][perm[j]];
  for (int j = 0; j < 5; ++j) {
    float Q[9];
    for (int a = 0; a < 3; ++a)
      for (int c = 0; c < 3; ++c) {
        float s = 0.f;
        for (int b = 0; b < 3; ++b) s += A2c[j][a*3+b] * R[b][c];
        Q[a*3+c] = s;
      }
    float S[9];
    for (int c = 0; c < 3; ++c)
      for (int dd = 0; dd < 3; ++dd) {
        float s = 0.f;
        for (int a = 0; a < 3; ++a) s += Q[a*3+c] * R[a][dd];
        S[c*3+dd] = s;
      }
    for (int i = 0; i < 5; ++i) {
      float s = 0.f;
      for (int c = 0; c < 3; ++c)
        for (int dd = 0; dd < 3; ++dd) s += S[c*3+dd] * A2c[i][c*3+dd];
      D2g[(size_t)e*25 + j*5 + i] = s;
    }
  }
}

// ---------------------------------------------------------------------------
// Edge embedding: [RBF(512) | send_emb | recv_emb] -> f16 (E x 768)
// ---------------------------------------------------------------------------
__global__ void k_eemb(const float* __restrict__ dist, const int* __restrict__ species,
                       const int* __restrict__ senders, const int* __restrict__ receivers,
                       const float* __restrict__ send_emb, const float* __restrict__ recv_emb,
                       _Float16* __restrict__ eemb, int E)
{
  int e = blockIdx.x;
  float d = dist[e];
  int ss = species[senders[e]];
  int rs = species[receivers[e]];
  const float sig = 2.f * CUTOFF_F / (float)(NRBF_ - 1);
  for (int idx = threadIdx.x; idx < EIN_; idx += blockDim.x) {
    float v;
    if (idx < NRBF_) {
      float mu = (float)idx * (CUTOFF_F / (float)(NRBF_ - 1));
      float t = (d - mu) / sig;
      v = __expf(-t * t);
    } else if (idx < NRBF_ + EC_DIM) {
      v = send_emb[ss * EC_DIM + (idx - NRBF_)];
    } else {
      v = recv_emb[rs * EC_DIM + (idx - NRBF_ - EC_DIM)];
    }
    eemb[(size_t)e * EIN_ + idx] = (_Float16)v;
  }
}

// ---------------------------------------------------------------------------
// x init: x[n,0,:] = sphere_emb[species[n]] + csd[0]; rows 1..8 = 0
// ---------------------------------------------------------------------------
__global__ void k_init_x(const int* __restrict__ species, const float* __restrict__ sph,
                         const float* __restrict__ csd, float* __restrict__ x, int N)
{
  int n = blockIdx.x, c = threadIdx.x;
  float* xp = x + (size_t)n * 9 * C_DIM;
  xp[c] = sph[species[n] * C_DIM + c] + csd[c];
  for (int l = 1; l < 9; ++l) xp[l * C_DIM + c] = 0.f;
}

// ---------------------------------------------------------------------------
// Async B-tile stage: copy 128 rows x 32 k of Wt (N,K layout) into a padded
// LDS tile using CDNA5 async global->LDS b128 copies (ASYNCcnt-tracked).
// Each of the 256 threads moves 32 contiguous bytes (2 x b128).
// ---------------------------------------------------------------------------
__device__ __forceinline__ void async_copy_btile(const _Float16* __restrict__ Wt,
                                                 _Float16 (*bsh)[48],
                                                 int n0, int k0, int K)
{
  int t = threadIdx.x;
  int nl  = t >> 1;          // LDS row (output column)
  int h32 = t & 1;           // which 32B half of the 64B row payload
  const _Float16* src = Wt + (size_t)(n0 + nl) * K + k0 + h32 * 16;
  _Float16* dst = &bsh[nl][h32 * 16];
  unsigned int       la = (unsigned int)(size_t)dst;        // low 32 bits = LDS offset
  unsigned long long ga = (unsigned long long)(size_t)src;
  asm volatile("global_load_async_to_lds_b128 %0, %1, off"
               :: "v"(la), "v"(ga) : "memory");
  asm volatile("global_load_async_to_lds_b128 %0, %1, off"
               :: "v"((unsigned int)(la + 16)), "v"((unsigned long long)(ga + 16)) : "memory");
}

// ---------------------------------------------------------------------------
// WMMA GEMM: C[M,N] = act(A[M,K](f16) @ W[K,N](f16) + bias),
// W given pre-transposed as Wt[N][K].
// block = 256 threads (8 waves); tile 128(M) x 128(N); K step 32;
// double-buffered async-LDS weight staging. ACT: 0=none,1=silu,2=sigmoid.
// Requires K%32==0, N%128==0.
// ---------------------------------------------------------------------------
template<int ACT, typename OutT>
__global__ __launch_bounds__(256)
void k_gemm(const _Float16* __restrict__ A, const _Float16* __restrict__ Wt,
            const float* __restrict__ bias, OutT* __restrict__ Cmat,
            int M, int K, int N)
{
  __shared__ alignas(32) _Float16 bsh[2][128][48];   // (n_local, kk), 96B row stride
  const int lane = threadIdx.x & 31;
  const int wave = threadIdx.x >> 5;
  const int half = lane >> 4;       // 0: lanes 0-15, 1: lanes 16-31
  const int lmod = lane & 15;
  const int m0 = blockIdx.x * 128;
  const int n0 = blockIdx.y * 128;
  int row  = m0 + wave * 16 + lmod;
  int rowc = (row < M) ? row : (M - 1);
  const _Float16* arow = A + (size_t)rowc * K;

  v8f acc[8];
#pragma unroll
  for (int j = 0; j < 8; ++j)
#pragma unroll
    for (int i = 0; i < 8; ++i) acc[j][i] = 0.f;

  const int nsteps = K >> 5;
  async_copy_btile(Wt, bsh[0], n0, 0, K);

  for (int s = 0; s < nsteps; ++s) {
    asm volatile("s_wait_asynccnt 0x0" ::: "memory");
    __syncthreads();                               // buffer s&1 is ready for all waves
    if (s + 1 < nsteps)
      async_copy_btile(Wt, bsh[(s + 1) & 1], n0, (s + 1) * 32, K);  // overlaps WMMAs below
    const int k0 = s * 32;
    if (k0 + 32 < K) __builtin_prefetch(arow + k0 + 32, 0, 3);
    // A fragment: elems 0..15 -> k0+half*16.. (contiguous, matches B)
    v16h af = *(const v16h*)(arow + k0 + half * 16);
    const _Float16 (*bb)[48] = bsh[s & 1];
#pragma unroll
    for (int j = 0; j < 8; ++j) {
      // B fragment: lane holds column n = j*16+lmod, k = k0 + half*16 + (0..15)
      v16h bf = *(const v16h*)(&bb[j * 16 + lmod][half * 16]);
      acc[j] = __builtin_amdgcn_wmma_f32_16x16x32_f16(
          false, af, false, bf, (short)0, acc[j], false, false);
    }
  }

#pragma unroll
  for (int j = 0; j < 8; ++j) {
    int n = n0 + j * 16 + lmod;
    float b = bias ? bias[n] : 0.f;
#pragma unroll
    for (int r = 0; r < 8; ++r) {
      int m = m0 + wave * 16 + half * 8 + r;   // C/D layout: VGPR r -> row r (+8 upper lanes)
      if (m < M) {
        float v = acc[j][r] + b;
        if (ACT == 1)      v = v / (1.f + __expf(-v));   // silu
        else if (ACT == 2) v = 1.f / (1.f + __expf(-v)); // sigmoid
        Cmat[(size_t)m * N + n] = (OutT)v;
      }
    }
  }
}

// ---------------------------------------------------------------------------
// zf scatter: x[recv] += rot_back(zf) * env / RESCALE  (zf nonzero rows 0,2,6)
// ---------------------------------------------------------------------------
__global__ void k_scatter_zf(const float* __restrict__ w3, const float* __restrict__ D1g,
                             const float* __restrict__ D2g, const float* __restrict__ env,
                             const int* __restrict__ receivers, float* __restrict__ x, int E)
{
  __shared__ float d1r[3], d2r[5];
  int e = blockIdx.x, c = threadIdx.x;
  if (c < 3) d1r[c] = D1g[(size_t)e*9  + 3  + c];  // D1 row j=1 (back-rot of row 2)
  if (c < 5) d2r[c] = D2g[(size_t)e*25 + 10 + c];  // D2 row j=2 (back-rot of row 6)
  __syncthreads();
  float en = env[e] / RESCALE_F;
  const float* wp = w3 + (size_t)e * 3 * C_DIM;
  float w0 = wp[c], w1 = wp[C_DIM + c], w2 = wp[2*C_DIM + c];
  float* xp = x + (size_t)receivers[e] * 9 * C_DIM + c;
  atomicAdd(&xp[0], en * w0);
  for (int i = 0; i < 3; ++i) atomicAdd(&xp[(1 + i) * C_DIM], en * d1r[i] * w1);
  for (int i = 0; i < 5; ++i) atomicAdd(&xp[(4 + i) * C_DIM], en * d2r[i] * w2);
}

// ---------------------------------------------------------------------------
// ln_sh: layernorm on l=0, RMS-norm over l=1..3 and l=4..8
// ---------------------------------------------------------------------------
__device__ __forceinline__ float blk_sum128(float v, float* sh)
{
  for (int o = 16; o > 0; o >>= 1) v += __shfl_down(v, o, 32);
  __syncthreads();
  if ((threadIdx.x & 31) == 0) sh[threadIdx.x >> 5] = v;
  __syncthreads();
  return sh[0] + sh[1] + sh[2] + sh[3];
}

__global__ void k_ln_sh(const float* __restrict__ x,
                        const float* __restrict__ g0, const float* __restrict__ b0,
                        const float* __restrict__ g1, const float* __restrict__ g2,
                        float* __restrict__ out, int N)
{
  __shared__ float sh[4];
  int n = blockIdx.x, c = threadIdx.x;
  const float* xp = x + (size_t)n * 9 * C_DIM;
  float* op = out + (size_t)n * 9 * C_DIM;
  float s = xp[c];
  float mean = blk_sum128(s, sh) * (1.f / 128.f);
  float dev = s - mean;
  float var = blk_sum128(dev * dev, sh) * (1.f / 128.f);
  op[c] = dev * rsqrtf(var + 1e-5f) * g0[c] + b0[c];
  float q1 = 0.f;
  for (int l = 1; l < 4; ++l) { float v = xp[l * C_DIM + c]; q1 += v * v; }
  float r1 = sqrtf(blk_sum128(q1, sh) * (1.f / 384.f) + 1e-5f);
  float ir1 = 1.f / r1;
  for (int l = 1; l < 4; ++l) op[l * C_DIM + c] = xp[l * C_DIM + c] * ir1 * g1[c];
  float q2 = 0.f;
  for (int l = 4; l < 9; ++l) { float v = xp[l * C_DIM + c]; q2 += v * v; }
  float r2 = sqrtf(blk_sum128(q2, sh) * (1.f / 640.f) + 1e-5f);
  float ir2 = 1.f / r2;
  for (int l = 4; l < 9; ++l) op[l * C_DIM + c] = xp[l * C_DIM + c] * ir2 * g2[c];
}

// ---------------------------------------------------------------------------
// gather + forward-rot for senders and receivers -> cat f16 (E, 9, 256)
// ---------------------------------------------------------------------------
__global__ void k_gather_rot_cat(const float* __restrict__ xn, const int* __restrict__ senders,
                                 const int* __restrict__ receivers, const float* __restrict__ D1g,
                                 const float* __restrict__ D2g, _Float16* __restrict__ cat, int E)
{
  __shared__ float d1[9], d2[25];
  int e = blockIdx.x, c = threadIdx.x;
  if (c < 9)  d1[c] = D1g[(size_t)e*9  + c];
  if (c < 25) d2[c] = D2g[(size_t)e*25 + c];
  __syncthreads();
  for (int side = 0; side < 2; ++side) {
    int node = (side == 0) ? senders[e] : receivers[e];
    const float* xp = xn + (size_t)node * 9 * C_DIM + c;
    float y[9];
    for (int l = 0; l < 9; ++l) y[l] = xp[l * C_DIM];
    float o[9];
    o[0] = y[0];
    for (int i = 0; i < 3; ++i)
      o[1 + i] = d1[i*3+0]*y[1] + d1[i*3+1]*y[2] + d1[i*3+2]*y[3];
    for (int i = 0; i < 5; ++i) {
      float s = 0.f;
      for (int j = 0; j < 5; ++j) s += d2[i*5+j] * y[4 + j];
      o[4 + i] = s;
    }
    _Float16* cp = cat + (size_t)e * 9 * 256 + side * C_DIM + c;
    for (int l = 0; l < 9; ++l) cp[l * 256] = (_Float16)o[l];
  }
}

// ---------------------------------------------------------------------------
// h *= rad (broadcast over l); emit s16 = f16(h[:,0,:])
// ---------------------------------------------------------------------------
__global__ void k_modulate(float* __restrict__ h, const float* __restrict__ radh,
                           _Float16* __restrict__ s16, int E)
{
  int e = blockIdx.x, c = threadIdx.x;
  float r = radh[(size_t)e * C_DIM + c];
  float* hp = h + (size_t)e * 9 * C_DIM + c;
  for (int l = 0; l < 9; ++l) hp[l * C_DIM] *= r;
  s16[(size_t)e * C_DIM + c] = (_Float16)hp[0];
}

// ---------------------------------------------------------------------------
// gating: l=0 silu, l=1..3 *g1, l=4..8 *g2; f16 out for w_out GEMM
// ---------------------------------------------------------------------------
__global__ void k_gate(const float* __restrict__ h, const float* __restrict__ g1,
                       const float* __restrict__ g2, _Float16* __restrict__ h2, int E)
{
  int e = blockIdx.x, c = threadIdx.x;
  float a = g1[(size_t)e * C_DIM + c];
  float b = g2[(size_t)e * C_DIM + c];
  const float* hp = h + (size_t)e * 9 * C_DIM + c;
  _Float16* op = h2 + (size_t)e * 9 * C_DIM + c;
  float s = hp[0];
  op[0] = (_Float16)(s / (1.f + __expf(-s)));
  for (int l = 1; l < 4; ++l) op[l * C_DIM] = (_Float16)(hp[l * C_DIM] * a);
  for (int l = 4; l < 9; ++l) op[l * C_DIM] = (_Float16)(hp[l * C_DIM] * b);
}

// ---------------------------------------------------------------------------
// rot_back(msg) * env, atomic scatter-add into x[receiver]
// ---------------------------------------------------------------------------
__global__ void k_scatter_msg(const float* __restrict__ msg, const float* __restrict__ D1g,
                              const float* __restrict__ D2g, const float* __restrict__ env,
                              const int* __restrict__ receivers, float* __restrict__ x, int E)
{
  __shared__ float d1[9], d2[25];
  int e = blockIdx.x, c = threadIdx.x;
  if (c < 9)  d1[c] = D1g[(size_t)e*9  + c];
  if (c < 25) d2[c] = D2g[(size_t)e*25 + c];
  __syncthreads();
  float en = env[e];
  const float* mp = msg + (size_t)e * 9 * C_DIM + c;
  float y[9];
  for (int l = 0; l < 9; ++l) y[l] = mp[l * C_DIM];
  float o[9];
  o[0] = y[0];
  for (int i = 0; i < 3; ++i)
    o[1 + i] = d1[0*3+i]*y[1] + d1[1*3+i]*y[2] + d1[2*3+i]*y[3];
  for (int i = 0; i < 5; ++i) {
    float s = 0.f;
    for (int j = 0; j < 5; ++j) s += d2[j*5+i] * y[4 + j];
    o[4 + i] = s;
  }
  float* xp = x + (size_t)receivers[e] * 9 * C_DIM + c;
  for (int l = 0; l < 9; ++l) atomicAdd(&xp[l * C_DIM], o[l] * en);
}

// ---------------------------------------------------------------------------
// grid projections
// ---------------------------------------------------------------------------
__global__ void k_to_grid(const float* __restrict__ xn, const float* __restrict__ TG,
                          _Float16* __restrict__ gout, int N)
{
  __shared__ float tg[NGRID_ * 9];
  int n = blockIdx.x, c = threadIdx.x;
  for (int idx = threadIdx.x; idx < NGRID_ * 9; idx += blockDim.x) tg[idx] = TG[idx];
  __syncthreads();
  const float* xp = xn + (size_t)n * 9 * C_DIM + c;
  float y[9];
  for (int l = 0; l < 9; ++l) y[l] = xp[l * C_DIM];
  _Float16* gp = gout + (size_t)n * NGRID_ * C_DIM + c;
  for (int p = 0; p < NGRID_; ++p) {
    float a = 0.f;
    for (int l = 0; l < 9; ++l) a += tg[p * 9 + l] * y[l];
    gp[p * C_DIM] = (_Float16)a;
  }
}

__global__ void k_from_grid_add(const float* __restrict__ t2, const float* __restrict__ FG,
                                float* __restrict__ x, int N)
{
  __shared__ float fg[9 * NGRID_];
  int n = blockIdx.x, c = threadIdx.x;
  for (int idx = threadIdx.x; idx < 9 * NGRID_; idx += blockDim.x) fg[idx] = FG[idx];
  __syncthreads();
  const float* tp = t2 + (size_t)n * NGRID_ * C_DIM + c;
  float a[9];
  for (int l = 0; l < 9; ++l) a[l] = 0.f;
  for (int p = 0; p < NGRID_; ++p) {
    float v = tp[p * C_DIM];
    for (int l = 0; l < 9; ++l) a[l] += fg[l * NGRID_ + p] * v;
  }
  float* xp = x + (size_t)n * 9 * C_DIM + c;
  for (int l = 0; l < 9; ++l) xp[l * C_DIM] += a[l];
}

// ---------------------------------------------------------------------------
// host side
// ---------------------------------------------------------------------------
#define GEMM(ACT_, OUTT_, A_, W_, B_, C_, M_, K_, N_) \
  k_gemm<ACT_, OUTT_><<<dim3(((M_) + 127) / 128, (N_) / 128), 256, 0, stream>>>( \
      A_, W_, B_, C_, M_, K_, N_)

extern "C" void kernel_launch(void* const* d_in, const int* in_sizes, int n_in,
                              void* d_out, int out_size, void* d_ws, size_t ws_size,
                              hipStream_t stream)
{
  (void)in_sizes; (void)n_in; (void)out_size; (void)ws_size;
  const int E = N_EDGES_, Nn = N_NODES_;

  // --- input pointers (top-level dict in setup_inputs() insertion order,
  //     'params' pytree leaves in alphabetical key order, lists in order) ---
  const float* edge_vectors = (const float*)d_in[0];
  const int*   node_species = (const int*)  d_in[1];
  const float* csd          = (const float*)d_in[2];  // (1, C)
  const int*   senders      = (const int*)  d_in[3];
  const int*   receivers    = (const int*)  d_in[4];
  /* d_in[5] = n_node (single graph, unused) */
  const float* rot_gamma    = (const float*)d_in[6];
  int pi = 7;
  const float *edeg_w[3], *edeg_b[3];
  for (int i = 0; i < 3; ++i) { edeg_w[i] = (const float*)d_in[pi++]; edeg_b[i] = (const float*)d_in[pi++]; }
  struct LayerP {
    const float *ff_w0, *ff_b0, *ff_w1, *ff_b1;
    const float *n1[4], *n2[4];
    const float *rad_w[3], *rad_b[3];
    const float *w_msg, *w_out, *wg1, *wg2;
  } L[NLAYERS_];
  for (int l = 0; l < NLAYERS_; ++l) {
    L[l].ff_w0 = (const float*)d_in[pi++]; L[l].ff_b0 = (const float*)d_in[pi++];
    L[l].ff_w1 = (const float*)d_in[pi++]; L[l].ff_b1 = (const float*)d_in[pi++];
    for (int i = 0; i < 4; ++i) L[l].n1[i] = (const float*)d_in[pi++];
    for (int i = 0; i < 4; ++i) L[l].n2[i] = (const float*)d_in[pi++];
    for (int i = 0; i < 3; ++i) { L[l].rad_w[i] = (const float*)d_in[pi++]; L[l].rad_b[i] = (const float*)d_in[pi++]; }
    L[l].w_msg = (const float*)d_in[pi++];
    L[l].w_out = (const float*)d_in[pi++];
    L[l].wg1   = (const float*)d_in[pi++];
    L[l].wg2   = (const float*)d_in[pi++];
  }
  const float* nf[4];
  for (int i = 0; i < 4; ++i) nf[i] = (const float*)d_in[pi++];
  const float* recv_emb   = (const float*)d_in[pi++];
  const float* send_emb   = (const float*)d_in[pi++];
  const float* sphere_emb = (const float*)d_in[pi++];

  // --- workspace bump allocator (256B aligned) ---
  char* wsb = (char*)d_ws;
  size_t off = 0;
  auto alloc = [&](size_t bytes) -> void* {
    void* p = wsb + off;
    off = (off + bytes + 255) & ~(size_t)255;
    return p;
  };

  // f16 weights, pre-transposed to (N, K)
  _Float16* edeg16[3];
  edeg16[0] = (_Float16*)alloc((size_t)EIN_ * EC_DIM * 2);
  edeg16[1] = (_Float16*)alloc((size_t)EC_DIM * EC_DIM * 2);
  edeg16[2] = (_Float16*)alloc((size_t)EC_DIM * 3 * C_DIM * 2);
  struct LayerW { _Float16 *rad[3], *wmsg, *wout, *wg1, *wg2, *ff0, *ff1; } LW[NLAYERS_];
  for (int l = 0; l < NLAYERS_; ++l) {
    LW[l].rad[0] = (_Float16*)alloc((size_t)EIN_ * EC_DIM * 2);
    LW[l].rad[1] = (_Float16*)alloc((size_t)EC_DIM * EC_DIM * 2);
    LW[l].rad[2] = (_Float16*)alloc((size_t)EC_DIM * H_DIM * 2);
    LW[l].wmsg = (_Float16*)alloc((size_t)2 * C_DIM * H_DIM * 2);
    LW[l].wout = (_Float16*)alloc((size_t)H_DIM * C_DIM * 2);
    LW[l].wg1  = (_Float16*)alloc((size_t)H_DIM * H_DIM * 2);
    LW[l].wg2  = (_Float16*)alloc((size_t)H_DIM * H_DIM * 2);
    LW[l].ff0  = (_Float16*)alloc((size_t)C_DIM * H_DIM * 2);
    LW[l].ff1  = (_Float16*)alloc((size_t)H_DIM * C_DIM * 2);
  }

  float* TG   = (float*)alloc((size_t)NGRID_ * 9 * 4);
  float* FG   = (float*)alloc((size_t)9 * NGRID_ * 4);
  float* D1g  = (float*)alloc((size_t)E * 9 * 4);
  float* D2g  = (float*)alloc((size_t)E * 25 * 4);
  float* envb = (float*)alloc((size_t)E * 4);
  float* dist = (float*)alloc((size_t)E * 4);
  _Float16* eemb = (_Float16*)alloc((size_t)E * EIN_ * 2);
  float* x    = (float*)alloc((size_t)Nn * 9 * C_DIM * 4);
  float* xn   = (float*)alloc((size_t)Nn * 9 * C_DIM * 4);
  _Float16* hid1 = (_Float16*)alloc((size_t)E * EC_DIM * 2);
  _Float16* hid2 = (_Float16*)alloc((size_t)E * EC_DIM * 2);
  float* w3   = (float*)alloc((size_t)E * 3 * C_DIM * 4);  // also reused as radh (E x H)
  _Float16* s16 = (_Float16*)alloc((size_t)E * C_DIM * 2);
  float* g1b  = (float*)alloc((size_t)E * H_DIM * 4);
  float* g2b  = (float*)alloc((size_t)E * H_DIM * 4);
  size_t szA = (size_t)E * 9 * 256 * 2;                            // cat ; grid16
  size_t szB = (size_t)Nn * NGRID_ * C_DIM * 4;                    // t2
  size_t szB2 = (size_t)E * 9 * C_DIM * 4;                         // h ; msg
  if (szB2 > szB) szB = szB2;
  size_t szC = (size_t)Nn * NGRID_ * C_DIM * 2;                    // t1
  size_t szC2 = (size_t)E * 9 * C_DIM * 2;                         // h2
  if (szC2 > szC) szC = szC2;
  char* arenaA = (char*)alloc(szA);
  char* arenaB = (char*)alloc(szB);
  char* arenaC = (char*)alloc(szC);
  _Float16* cat    = (_Float16*)arenaA;
  _Float16* grid16 = (_Float16*)arenaA;
  float*    h      = (float*)arenaB;
  float*    msg    = (float*)arenaB;   // h dead before msg GEMM writes
  float*    t2     = (float*)arenaB;   // msg dead before FF
  _Float16* h2     = (_Float16*)arenaC;
  _Float16* t1     = (_Float16*)arenaC; // h2 dead before ff0
  float*    radh   = w3;               // edeg output dead before layers

  auto convT = [&](const float* s, _Float16* d, int K, int N) {
    k_transpose_f16<<<dim3((K * N + 255) / 256), 256, 0, stream>>>(s, d, K, N);
  };

  // --- weight conversion to transposed f16 ---
  convT(edeg_w[0], edeg16[0], EIN_, EC_DIM);
  convT(edeg_w[1], edeg16[1], EC_DIM, EC_DIM);
  convT(edeg_w[2], edeg16[2], EC_DIM, 3 * C_DIM);
  for (int l = 0; l < NLAYERS_; ++l) {
    convT(L[l].rad_w[0], LW[l].rad[0], EIN_, EC_DIM);
    convT(L[l].rad_w[1], LW[l].rad[1], EC_DIM, EC_DIM);
    convT(L[l].rad_w[2], LW[l].rad[2], EC_DIM, H_DIM);
    convT(L[l].w_msg, LW[l].wmsg, 2 * C_DIM, H_DIM);
    convT(L[l].w_out, LW[l].wout, H_DIM, C_DIM);
    convT(L[l].wg1,   LW[l].wg1,  H_DIM, H_DIM);
    convT(L[l].wg2,   LW[l].wg2,  H_DIM, H_DIM);
    convT(L[l].ff_w0, LW[l].ff0,  C_DIM, H_DIM);
    convT(L[l].ff_w1, LW[l].ff1,  H_DIM, C_DIM);
  }

  // --- preamble ---
  k_make_grid<<<1, 128, 0, stream>>>(TG, FG);
  k_edge_geom<<<dim3((E + 255) / 256), 256, 0, stream>>>(edge_vectors, rot_gamma, D1g, D2g, envb, dist, E);
  k_eemb<<<dim3(E), 256, 0, stream>>>(dist, node_species, senders, receivers, send_emb, recv_emb, eemb, E);
  k_init_x<<<dim3(Nn), 128, 0, stream>>>(node_species, sphere_emb, csd, x, Nn);

  // --- edge-degree embedding MLP + zf scatter ---
  GEMM(1, _Float16, eemb, edeg16[0], edeg_b[0], hid1, E, EIN_, EC_DIM);
  GEMM(1, _Float16, hid1, edeg16[1], edeg_b[1], hid2, E, EC_DIM, EC_DIM);
  GEMM(0, float,    hid2, edeg16[2], edeg_b[2], w3,   E, EC_DIM, 3 * C_DIM);
  k_scatter_zf<<<dim3(E), 128, 0, stream>>>(w3, D1g, D2g, envb, receivers, x, E);

  // --- layers ---
  for (int l = 0; l < NLAYERS_; ++l) {
    k_ln_sh<<<dim3(Nn), 128, 0, stream>>>(x, L[l].n1[0], L[l].n1[1], L[l].n1[2], L[l].n1[3], xn, Nn);
    k_gather_rot_cat<<<dim3(E), 128, 0, stream>>>(xn, senders, receivers, D1g, D2g, cat, E);
    GEMM(1, _Float16, eemb, LW[l].rad[0], L[l].rad_b[0], hid1, E, EIN_, EC_DIM);
    GEMM(1, _Float16, hid1, LW[l].rad[1], L[l].rad_b[1], hid2, E, EC_DIM, EC_DIM);
    GEMM(0, float,    hid2, LW[l].rad[2], L[l].rad_b[2], radh, E, EC_DIM, H_DIM);
    GEMM(0, float, cat, LW[l].wmsg, (const float*)nullptr, h, E * 9, 2 * C_DIM, H_DIM);
    k_modulate<<<dim3(E), 128, 0, stream>>>(h, radh, s16, E);
    GEMM(2, float, s16, LW[l].wg1, (const float*)nullptr, g1b, E, H_DIM, H_DIM);
    GEMM(2, float, s16, LW[l].wg2, (const float*)nullptr, g2b, E, H_DIM, H_DIM);
    k_gate<<<dim3(E), 128, 0, stream>>>(h, g1b, g2b, h2, E);
    GEMM(0, float, h2, LW[l].wout, (const float*)nullptr, msg, E * 9, H_DIM, C_DIM);
    k_scatter_msg<<<dim3(E), 128, 0, stream>>>(msg, D1g, D2g, envb, receivers, x, E);
    k_ln_sh<<<dim3(Nn), 128, 0, stream>>>(x, L[l].n2[0], L[l].n2[1], L[l].n2[2], L[l].n2[3], xn, Nn);
    k_to_grid<<<dim3(Nn), 128, 0, stream>>>(xn, TG, grid16, Nn);
    GEMM(1, _Float16, grid16, LW[l].ff0, L[l].ff_b0, t1, Nn * NGRID_, C_DIM, H_DIM);
    GEMM(0, float,    t1,     LW[l].ff1, L[l].ff_b1, t2, Nn * NGRID_, H_DIM, C_DIM);
    k_from_grid_add<<<dim3(Nn), 128, 0, stream>>>(t2, FG, x, Nn);
  }

  // --- final layernorm -> d_out ---
  k_ln_sh<<<dim3(Nn), 128, 0, stream>>>(x, nf[0], nf[1], nf[2], nf[3], (float*)d_out, Nn);
}
